// GraphSAGE_35072702939338
// MI455X (gfx1250) — matrix operand order repaired and checked by reading, not compiled, not run on hardware
//
#include <hip/hip_runtime.h>

// ---------------------------------------------------------------------------
// GraphSAGE 2-layer forward for MI455X (gfx1250, wave32, WMMA)
// ---------------------------------------------------------------------------

typedef __bf16 bf16_t;
typedef __attribute__((ext_vector_type(16))) __bf16 v16bf;
typedef __attribute__((ext_vector_type(8)))  float  v8f;

#define NNODES 50000
#define NEDGES 800000
#define DIN    128
#define DH     256
#define DOUT   64

// ------------------------------- utilities --------------------------------

__global__ __launch_bounds__(256) void zero_f32(float* p, int n) {
    int i = blockIdx.x * blockDim.x + threadIdx.x;
    int stride = gridDim.x * blockDim.x;
    for (; i < n; i += stride) p[i] = 0.0f;
}

// Convert all four weight matrices f32 -> bf16 (98304 elements total).
__global__ __launch_bounds__(256) void cvt_weights(
        const float* W1l, const float* W1r, const float* W2l, const float* W2r,
        bf16_t* o1l, bf16_t* o1r, bf16_t* o2l, bf16_t* o2r) {
    int i = blockIdx.x * blockDim.x + threadIdx.x;
    if (i < 32768)        o1l[i]         = (bf16_t)W1l[i];
    else if (i < 65536)   o1r[i - 32768] = (bf16_t)W1r[i - 32768];
    else if (i < 81920)   o2l[i - 65536] = (bf16_t)W2l[i - 65536];
    else if (i < 98304)   o2r[i - 81920] = (bf16_t)W2r[i - 81920];
}

// --------------------------- scatter (segment sum) ------------------------

// One wave per edge; lane handles 4 contiguous features of x[src] -> msg[dst].
// Lane 0 also accumulates the destination degree.
__global__ __launch_bounds__(256) void scatter1(const long long* __restrict__ ei,
                                                const float* __restrict__ x,
                                                float* __restrict__ msg,
                                                float* __restrict__ deg) {
    unsigned gid  = blockIdx.x * blockDim.x + threadIdx.x;
    unsigned edge = gid >> 5;
    unsigned lane = gid & 31u;
    if (edge >= NEDGES) return;
    long long s = ei[edge];
    long long d = ei[NEDGES + edge];
    const float4 v = *(const float4*)(x + (size_t)s * DIN + lane * 4);
    float* o = msg + (size_t)d * DIN + lane * 4;
    atomicAdd(o + 0, v.x);
    atomicAdd(o + 1, v.y);
    atomicAdd(o + 2, v.z);
    atomicAdd(o + 3, v.w);
    if (lane == 0) atomicAdd(deg + d, 1.0f);
}

// One wave per edge; lane handles 8 bf16 features of h[src] -> f32 msg[dst].
__global__ __launch_bounds__(256) void scatter2(const long long* __restrict__ ei,
                                                const bf16_t* __restrict__ h,
                                                float* __restrict__ msg) {
    unsigned gid  = blockIdx.x * blockDim.x + threadIdx.x;
    unsigned edge = gid >> 5;
    unsigned lane = gid & 31u;
    if (edge >= NEDGES) return;
    long long s = ei[edge];
    long long d = ei[NEDGES + edge];
    const bf16_t* hr = h + (size_t)s * DH + lane * 8;
    float* o = msg + (size_t)d * DH + lane * 8;
#pragma unroll
    for (int i = 0; i < 8; i++) atomicAdd(o + i, (float)hr[i]);
}

// ------------------------- WMMA fragment loaders --------------------------
// Fragment layouts used here (wave32; contiguous K for A and B, not the ISA table order):
//  A (16x32 bf16): lane L -> row M = L%16; elements 0..7  -> K = (L/16)*16 + e
//                                          elements 8..15 -> K = (L/16)*16 + e
//  B (32x16 bf16): lane L -> col N = L%16; element e      -> K = (L/16)*16 + e
//  C/D (16x16 f32): lane L -> col N = L%16; VGPR r -> row M = (L/16)*8 + r

// A fragment from an f32 row (with per-row scale), K-chunk base already applied.
__device__ __forceinline__ v16bf frag_a_f32(const float* rowk, unsigned half, float scale) {
    const float* p0 = rowk + half * 16;     // K = half*16 .. +7   (32B contiguous)
    const float* p1 = p0 + 8;               // K = half*16+8 .. +7
    v16bf r;
#pragma unroll
    for (int i = 0; i < 8; i++) r[i] = (bf16_t)(p0[i] * scale);
#pragma unroll
    for (int i = 0; i < 8; i++) r[8 + i] = (bf16_t)(p1[i] * scale);
    return r;
}

// A fragment from a bf16 row.
__device__ __forceinline__ v16bf frag_a_bf16(const bf16_t* rowk, unsigned half) {
    const bf16_t* p0 = rowk + half * 16;
    const bf16_t* p1 = p0 + 8;
    v16bf r;
#pragma unroll
    for (int i = 0; i < 8; i++) r[i] = p0[i];
#pragma unroll
    for (int i = 0; i < 8; i++) r[8 + i] = p1[i];
    return r;
}

// ------------------------------- GEMM layer 1 -----------------------------
// h[50000,256] = relu( (msg/deg) @ W1l^T + x @ W1r^T + b1 ), output bf16.
// One wave computes one 16x16 tile. Mtiles=3125 (exact), Ntiles=16.
__global__ __launch_bounds__(256) void gemm1(const float* __restrict__ msg,
                                             const float* __restrict__ deg,
                                             const float* __restrict__ x,
                                             const bf16_t* __restrict__ Wl,
                                             const bf16_t* __restrict__ Wr,
                                             const float* __restrict__ b1,
                                             bf16_t* __restrict__ h) {
    unsigned wave = (blockIdx.x * blockDim.x + threadIdx.x) >> 5;   // tile id
    unsigned lane = threadIdx.x & 31u;
    unsigned mt = wave >> 4, nt = wave & 15u;
    unsigned half = lane >> 4, ml = lane & 15u;
    unsigned row = mt * 16 + ml;          // A-operand row this lane loads
    unsigned n   = nt * 16 + ml;          // B column / output column
    float inv = 1.0f / fmaxf(deg[row], 1.0f);
    const float*  arow  = msg + (size_t)row * DIN;
    const float*  xrow  = x   + (size_t)row * DIN;
    const bf16_t* wlrow = Wl + (size_t)n * DIN + half * 16;   // B col n = W row n
    const bf16_t* wrrow = Wr + (size_t)n * DIN + half * 16;
    v8f acc = {};
#pragma unroll
    for (int kt = 0; kt < DIN / 32; kt++) {
        v16bf a1 = frag_a_f32(arow + kt * 32, half, inv);
        v16bf bl = *(const v16bf*)(wlrow + kt * 32);
        acc = __builtin_amdgcn_wmma_f32_16x16x32_bf16(false, a1, false, bl,
                                                      (short)0, acc, false, false);
        v16bf a2 = frag_a_f32(xrow + kt * 32, half, 1.0f);
        v16bf br = *(const v16bf*)(wrrow + kt * 32);
        acc = __builtin_amdgcn_wmma_f32_16x16x32_bf16(false, a2, false, br,
                                                      (short)0, acc, false, false);
    }
    float bias = b1[n];
#pragma unroll
    for (int r = 0; r < 8; r++) {
        unsigned m = mt * 16 + half * 8 + r;
        float v = fmaxf(acc[r] + bias, 0.0f);
        h[(size_t)m * DH + n] = (bf16_t)v;
    }
}

// ------------------------------- GEMM layer 2 -----------------------------
// out[50000,64] = (msg2/deg) @ W2l^T + h @ W2r^T + b2, f32 output.
// Mtiles=3125, Ntiles=4 -> 12500 tiles.
__global__ __launch_bounds__(256) void gemm2(const float* __restrict__ msg,
                                             const float* __restrict__ deg,
                                             const bf16_t* __restrict__ h,
                                             const bf16_t* __restrict__ Wl,
                                             const bf16_t* __restrict__ Wr,
                                             const float* __restrict__ b2,
                                             float* __restrict__ out) {
    unsigned wave = (blockIdx.x * blockDim.x + threadIdx.x) >> 5;
    if (wave >= 12500) return;            // wave-uniform guard (EXEC stays full)
    unsigned lane = threadIdx.x & 31u;
    unsigned mt = wave >> 2, nt = wave & 3u;
    unsigned half = lane >> 4, ml = lane & 15u;
    unsigned row = mt * 16 + ml;
    unsigned n   = nt * 16 + ml;
    float inv = 1.0f / fmaxf(deg[row], 1.0f);
    const float*  arow  = msg + (size_t)row * DH;
    const bf16_t* hrow  = h   + (size_t)row * DH;
    const bf16_t* wlrow = Wl + (size_t)n * DH + half * 16;
    const bf16_t* wrrow = Wr + (size_t)n * DH + half * 16;
    v8f acc = {};
#pragma unroll
    for (int kt = 0; kt < DH / 32; kt++) {
        v16bf a1 = frag_a_f32(arow + kt * 32, half, inv);
        v16bf bl = *(const v16bf*)(wlrow + kt * 32);
        acc = __builtin_amdgcn_wmma_f32_16x16x32_bf16(false, a1, false, bl,
                                                      (short)0, acc, false, false);
        v16bf a2 = frag_a_bf16(hrow + kt * 32, half);
        v16bf br = *(const v16bf*)(wrrow + kt * 32);
        acc = __builtin_amdgcn_wmma_f32_16x16x32_bf16(false, a2, false, br,
                                                      (short)0, acc, false, false);
    }
    float bias = b2[n];
#pragma unroll
    for (int r = 0; r < 8; r++) {
        unsigned m = mt * 16 + half * 8 + r;
        out[(size_t)m * DOUT + n] = acc[r] + bias;
    }
}

// ------------------------------- launcher ---------------------------------

extern "C" void kernel_launch(void* const* d_in, const int* in_sizes, int n_in,
                              void* d_out, int out_size, void* d_ws, size_t ws_size,
                              hipStream_t stream) {
    const float*     x   = (const float*)d_in[0];
    const long long* ei  = (const long long*)d_in[1];   // [2, NEDGES] int64
    const float*     W1l = (const float*)d_in[2];
    const float*     W1r = (const float*)d_in[3];
    const float*     b1  = (const float*)d_in[4];
    const float*     W2l = (const float*)d_in[5];
    const float*     W2r = (const float*)d_in[6];
    const float*     b2  = (const float*)d_in[7];
    float* out = (float*)d_out;

    // Workspace layout (bytes). msgsum2 (51.2MB) overlays msgsum1 (25.6MB):
    //   [0, 262144)                      : deg (50000 f32)
    //   [262144, +51,200,000)            : msg  (f32; layer1 uses first 25.6MB)
    //   [.., +25,600,000)                : h    (50000 x 256 bf16)
    //   [.., +196,608)                   : bf16 weights (W1l|W1r|W2l|W2r)
    char* ws = (char*)d_ws;
    float*  deg  = (float*)(ws);
    float*  msg  = (float*)(ws + 262144);
    bf16_t* h    = (bf16_t*)(ws + 262144 + 51200000);
    bf16_t* wbuf = (bf16_t*)(ws + 262144 + 51200000 + 25600000);
    bf16_t* w1l = wbuf;
    bf16_t* w1r = wbuf + 32768;
    bf16_t* w2l = wbuf + 65536;
    bf16_t* w2r = wbuf + 81920;

    // Layer 1
    zero_f32<<<256, 256, 0, stream>>>(deg, NNODES);
    zero_f32<<<2048, 256, 0, stream>>>(msg, NNODES * DIN);
    cvt_weights<<<384, 256, 0, stream>>>(W1l, W1r, W2l, W2r, w1l, w1r, w2l, w2r);
    scatter1<<<(NEDGES * 32) / 256, 256, 0, stream>>>(ei, x, msg, deg);
    gemm1<<<(NNODES / 16) * 16 / 8, 256, 0, stream>>>(msg, deg, x, w1l, w1r, b1, h);

    // Layer 2 (msg buffer reused after gemm1 has consumed it)
    zero_f32<<<2048, 256, 0, stream>>>(msg, NNODES * DH);
    scatter2<<<(NEDGES * 32) / 256, 256, 0, stream>>>(ei, h, msg);
    gemm2<<<1563, 256, 0, stream>>>(msg, deg, h, w2l, w2r, b2, out);
}
